// AttentionQV0_29850022707729
// MI455X (gfx1250) — matrix operand order repaired and checked by reading, not compiled, not run on hardware
//
#include <hip/hip_runtime.h>
#include <hip/hip_bf16.h>

// ---------------------------------------------------------------------------
// Problem constants (match reference)
// ---------------------------------------------------------------------------
#define N_TOK   1024
#define DIMX    256
#define HEADSX  8
#define DHX     64
#define INNERX  512
#define GBATCH  32          // b*p*heads
#define CCH     1024        // conv channels (=32*32)
#define HWX     32

typedef __attribute__((ext_vector_type(16))) __bf16 v16bf;
typedef __attribute__((ext_vector_type(8)))  float  v8f;

union FragBF { uint4 q[2]; v16bf v; };

__device__ __forceinline__ unsigned short f2bf(float f) {
  unsigned u = __float_as_uint(f);
  u += 0x7FFFu + ((u >> 16) & 1u);     // round-to-nearest-even
  return (unsigned short)(u >> 16);
}

// ---------------------------------------------------------------------------
// Register-blocked WMMA GEMM (direct-from-global operands).
// Fragment mapping per ISA 7.12.2:
//   A lane l: row=l&15, elems 0..7 <- K=k0+kh*8.., 8..15 <- K=k0+16+kh*8..
//   B lane l: col=l&15, elems 0..7 <- K=k0+kh*8.., 8..15 <- K=k0+16+kh*8..
// ---------------------------------------------------------------------------
template <int MT, int NT>
__device__ __forceinline__ void wmma_block(const unsigned short* __restrict__ A, int lda,
                                           const unsigned short* __restrict__ Bt, int ldb,
                                           int K, v8f acc[MT][NT]) {
  const int lane = threadIdx.x & 31;
  const int row  = lane & 15;
  const int kh   = lane >> 4;
  const unsigned short* ap[MT];
  const unsigned short* bp[NT];
#pragma unroll
  for (int i = 0; i < MT; i++) ap[i] = A  + (size_t)(i * 16 + row) * lda;
#pragma unroll
  for (int j = 0; j < NT; j++) bp[j] = Bt + (size_t)(j * 16 + row) * ldb;

  for (int k0 = 0; k0 < K; k0 += 32) {
    FragBF a[MT], b[NT];
#pragma unroll
    for (int i = 0; i < MT; i++) {
      a[i].q[0] = *(const uint4*)(ap[i] + k0 + kh * 8);
      a[i].q[1] = *(const uint4*)(ap[i] + k0 + 16 + kh * 8);
    }
#pragma unroll
    for (int j = 0; j < NT; j++) {
      b[j].q[0] = *(const uint4*)(bp[j] + k0 + kh * 8);
      b[j].q[1] = *(const uint4*)(bp[j] + k0 + 16 + kh * 8);
    }
    if (k0 + 32 < K) {
      __builtin_prefetch(ap[0] + k0 + 64, 0, 0);   // global_prefetch_b8
      __builtin_prefetch(bp[0] + k0 + 64, 0, 0);
    }
#pragma unroll
    for (int i = 0; i < MT; i++)
#pragma unroll
      for (int j = 0; j < NT; j++)
        acc[i][j] = __builtin_amdgcn_wmma_f32_16x16x32_bf16(
            false, a[i].v, false, b[j].v, (short)0, acc[i][j], false, false);
  }
}

// Block geometry: 4x2 tiles = 64x32 output per wave
#define BM 4
#define BN 2

// ---------------------------------------------------------------------------
// Weight prep: f32 -> bf16 (optionally transposed)
// ---------------------------------------------------------------------------
__global__ void k_transpose_bf16(unsigned short* __restrict__ dst,
                                 const float* __restrict__ src, int K, int N) {
  // src[K][N] -> dst[N][K]
  size_t id = (size_t)blockIdx.x * blockDim.x + threadIdx.x;
  if (id >= (size_t)K * N) return;
  int nrow = (int)(id / K);
  int kcol = (int)(id % K);
  dst[id] = f2bf(src[(size_t)kcol * N + nrow]);
}

__global__ void k_convert_bf16(unsigned short* __restrict__ dst,
                               const float* __restrict__ src, int n) {
  int id = blockIdx.x * blockDim.x + threadIdx.x;
  if (id < n) dst[id] = f2bf(src[id]);
}

// ---------------------------------------------------------------------------
// LayerNorm over DIM=256, one wave per row, bf16 output
// ---------------------------------------------------------------------------
__global__ void k_layernorm(const float* __restrict__ x,
                            const float* __restrict__ g,
                            const float* __restrict__ b,
                            unsigned short* __restrict__ xn) {
  int wave = threadIdx.x >> 5, lane = threadIdx.x & 31;
  int row = blockIdx.x * (blockDim.x >> 5) + wave;
  if (row >= 4096) return;
  const float4* xr = (const float4*)(x + (size_t)row * DIMX);
  float4 v0 = xr[lane * 2], v1 = xr[lane * 2 + 1];
  float s = v0.x + v0.y + v0.z + v0.w + v1.x + v1.y + v1.z + v1.w;
  for (int o = 16; o >= 1; o >>= 1) s += __shfl_xor(s, o, 32);
  float mu = s * (1.0f / 256.0f);
  float qsum = 0.f;
  { float d;
    d = v0.x - mu; qsum += d * d;  d = v0.y - mu; qsum += d * d;
    d = v0.z - mu; qsum += d * d;  d = v0.w - mu; qsum += d * d;
    d = v1.x - mu; qsum += d * d;  d = v1.y - mu; qsum += d * d;
    d = v1.z - mu; qsum += d * d;  d = v1.w - mu; qsum += d * d; }
  for (int o = 16; o >= 1; o >>= 1) qsum += __shfl_xor(qsum, o, 32);
  float rs = rsqrtf(qsum * (1.0f / 256.0f) + 1e-5f);
  float4 g0 = ((const float4*)g)[lane * 2], g1 = ((const float4*)g)[lane * 2 + 1];
  float4 b0 = ((const float4*)b)[lane * 2], b1 = ((const float4*)b)[lane * 2 + 1];
  unsigned short h[8];
  h[0] = f2bf((v0.x - mu) * rs * g0.x + b0.x);
  h[1] = f2bf((v0.y - mu) * rs * g0.y + b0.y);
  h[2] = f2bf((v0.z - mu) * rs * g0.z + b0.z);
  h[3] = f2bf((v0.w - mu) * rs * g0.w + b0.w);
  h[4] = f2bf((v1.x - mu) * rs * g1.x + b1.x);
  h[5] = f2bf((v1.y - mu) * rs * g1.y + b1.y);
  h[6] = f2bf((v1.z - mu) * rs * g1.z + b1.z);
  h[7] = f2bf((v1.w - mu) * rs * g1.w + b1.w);
  uint4 o4;
  o4.x = (unsigned)h[0] | ((unsigned)h[1] << 16);
  o4.y = (unsigned)h[2] | ((unsigned)h[3] << 16);
  o4.z = (unsigned)h[4] | ((unsigned)h[5] << 16);
  o4.w = (unsigned)h[6] | ((unsigned)h[7] << 16);
  ((uint4*)(xn + (size_t)row * DIMX))[lane] = o4;
}

// ---------------------------------------------------------------------------
// QKV GEMM: [4096 x 256] x [256 x 1536]; scatter to head-major q,k and v^T
// ---------------------------------------------------------------------------
__global__ void k_gemm_qkv(const unsigned short* __restrict__ xn,
                           const unsigned short* __restrict__ wT,
                           unsigned short* __restrict__ qb,
                           unsigned short* __restrict__ kb,
                           unsigned short* __restrict__ vT) {
  const int BLK_N = 48;   // 1536/32
  int w = blockIdx.x * (blockDim.x >> 5) + (threadIdx.x >> 5);
  if (w >= 64 * BLK_N) return;
  int bm = w / BLK_N, bn = w % BLK_N;
  v8f acc[BM][BN] = {};
  wmma_block<BM, BN>(xn + (size_t)bm * 64 * DIMX, DIMX,
                     wT + (size_t)bn * 32 * DIMX, DIMX, DIMX, acc);
  int lane = threadIdx.x & 31, row = lane & 15, kh = lane >> 4;
#pragma unroll
  for (int i = 0; i < BM; i++)
#pragma unroll
    for (int j = 0; j < BN; j++)
      for (int r = 0; r < 8; r++) {
        int M  = bm * 64 + i * 16 + r + 8 * kh;   // token row 0..4095
        int Nc = bn * 32 + j * 16 + row;          // qkv column 0..1535
        unsigned short hv = f2bf(acc[i][j][r]);
        int sect = Nc >> 9, jj = Nc & 511, h = jj >> 6, d = jj & 63;
        int g = ((M >> 10) << 3) + h, n = M & 1023;
        if (sect == 0)      qb[((size_t)g * N_TOK + n) * DHX + d] = hv;
        else if (sect == 1) kb[((size_t)g * N_TOK + n) * DHX + d] = hv;
        else                vT[((size_t)g * DHX + d) * N_TOK + n] = hv;
      }
}

// ---------------------------------------------------------------------------
// dots = q @ k^T * DH^-0.5   -> f32 [g][n][m]; per g: 16 x 32 blocks
// ---------------------------------------------------------------------------
__global__ void k_gemm_dots(const unsigned short* __restrict__ qb,
                            const unsigned short* __restrict__ kb,
                            float* __restrict__ dots) {
  int w = blockIdx.x * (blockDim.x >> 5) + (threadIdx.x >> 5);
  if (w >= GBATCH * 16 * 32) return;
  int g = w >> 9, rem = w & 511, bm = rem >> 5, bn = rem & 31;
  v8f acc[BM][BN] = {};
  wmma_block<BM, BN>(qb + ((size_t)g * N_TOK + bm * 64) * DHX, DHX,
                     kb + ((size_t)g * N_TOK + bn * 32) * DHX, DHX, DHX, acc);
  int lane = threadIdx.x & 31, row = lane & 15, kh = lane >> 4;
  float* base = dots + ((size_t)g << 20);
#pragma unroll
  for (int i = 0; i < BM; i++)
#pragma unroll
    for (int j = 0; j < BN; j++)
      for (int r = 0; r < 8; r++) {
        int n = bm * 64 + i * 16 + r + 8 * kh;
        int m = bn * 32 + j * 16 + row;
        base[(size_t)n * N_TOK + m] = acc[i][j][r] * 0.125f;  // DH^-0.5
      }
}

// ---------------------------------------------------------------------------
// Fused depthwise (1,2) conv_w + (2,1) conv_h with reflection pad.
// Reads dots in score layout, writes conv image transposed: t[g][spatial][chan]
// ---------------------------------------------------------------------------
__global__ void k_conv(const float* __restrict__ dots,
                       const float* __restrict__ w1,
                       const float* __restrict__ w2,
                       unsigned short* __restrict__ tb) {
  size_t id = (size_t)blockIdx.x * blockDim.x + threadIdx.x;
  if (id >= (size_t)GBATCH * CCH * HWX * HWX) return;
  int g = (int)(id >> 20), rem = (int)(id & 0xFFFFF);
  int c = rem >> 10, s = rem & 1023;
  int i = s >> 5, j = s & 31;
  int h1 = c >> 5, h2 = c & 31;
  int jp = (j == 31) ? 30 : j + 1;
  int ip = (i == 31) ? 30 : i + 1;
  const float* base = dots + ((size_t)g << 20);
  int r0 = (h1 << 5) + i, r1 = (h1 << 5) + ip;
  int c0 = (h2 << 5) + j, c1 = (h2 << 5) + jp;
  float a00 = base[(size_t)r0 * N_TOK + c0];
  float a01 = base[(size_t)r0 * N_TOK + c1];
  float a10 = base[(size_t)r1 * N_TOK + c0];
  float a11 = base[(size_t)r1 * N_TOK + c1];
  float w10 = w1[c * 2], w11 = w1[c * 2 + 1];
  float w20 = w2[c * 2], w21 = w2[c * 2 + 1];
  float u0 = a00 * w10 + a01 * w11;
  float u1 = a10 * w10 + a11 * w11;
  tb[((size_t)g << 20) + (size_t)s * CCH + c] = f2bf(u0 * w20 + u1 * w21);
}

// ---------------------------------------------------------------------------
// 1x1 conv GEMM (dominant): LDS-tiled, double-buffered with CDNA5
// global_load_async_to_lds_b128 (ASYNCcnt). Block = 256 threads computes
// 128x128 output (2x4 waves of 64x32). Per K-step stages 128x32 A-slab and
// 128x32 B-slab in LDS; fragments read back as ds_load_b128.
// Epilogue: BatchNorm + SiLU, scatter back to score layout.
// ---------------------------------------------------------------------------
__global__ void __launch_bounds__(256) k_gemm_mix(
    const unsigned short* __restrict__ w1x1b,
    const unsigned short* __restrict__ tb,
    const float* __restrict__ bng,
    const float* __restrict__ bnb,
    const float* __restrict__ bnm,
    const float* __restrict__ bnv,
    float* __restrict__ scores) {
  __shared__ unsigned short sA[2][128][32];   // [buf][row in M-slab][k]
  __shared__ unsigned short sB[2][128][32];   // [buf][col in N-slab][k]

  int blk = blockIdx.x;                 // GBATCH * 8 * 8 blocks
  int g = blk >> 6, rem = blk & 63;
  int bm = rem >> 3, bn = rem & 7;
  const unsigned short* Abase = w1x1b + (size_t)bm * 128 * CCH;
  const unsigned short* Bbase = tb + ((size_t)g << 20) + (size_t)bn * 128 * CCH;

  int t = threadIdx.x;
  int wave = t >> 5, lane = t & 31, row = lane & 15, kh = lane >> 4;
  int wm = wave >> 2, wn = wave & 3;

  // Each thread moves 2 A-chunks + 2 B-chunks (16B each) per K-step.
  // Chunk c in [0,512): slab row = c>>2, k-chunk = c&3 (8 bf16 per chunk).
  int c0 = t, c1 = t + 256;

  auto issue = [&](int buf, int k0) {
#pragma unroll
    for (int cc = 0; cc < 2; cc++) {
      int c = cc ? c1 : c0;
      int ar = c >> 2, kc = c & 3;
      unsigned la = (unsigned)(size_t)(&sA[buf][ar][kc * 8]);
      const unsigned short* ga = Abase + (size_t)ar * CCH + k0 + kc * 8;
      asm volatile("global_load_async_to_lds_b128 %0, %1, off"
                   :: "v"(la), "v"(ga) : "memory");
      unsigned lb = (unsigned)(size_t)(&sB[buf][ar][kc * 8]);
      const unsigned short* gb = Bbase + (size_t)ar * CCH + k0 + kc * 8;
      asm volatile("global_load_async_to_lds_b128 %0, %1, off"
                   :: "v"(lb), "v"(gb) : "memory");
    }
  };

  v8f acc[BM][BN] = {};

  issue(0, 0);
  asm volatile("s_wait_asynccnt 0x0" ::: "memory");
  __syncthreads();

  for (int ks = 0; ks < 32; ks++) {
    int cur = ks & 1;
    if (ks + 1 < 32) issue(cur ^ 1, (ks + 1) * 32);

    FragBF a[BM], b[BN];
#pragma unroll
    for (int i = 0; i < BM; i++) {
      const unsigned short* ar = &sA[cur][wm * 64 + i * 16 + row][0];
      a[i].q[0] = *(const uint4*)(ar + kh * 8);
      a[i].q[1] = *(const uint4*)(ar + 16 + kh * 8);
    }
#pragma unroll
    for (int j = 0; j < BN; j++) {
      const unsigned short* br = &sB[cur][wn * 32 + j * 16 + row][0];
      b[j].q[0] = *(const uint4*)(br + kh * 8);
      b[j].q[1] = *(const uint4*)(br + 16 + kh * 8);
    }
#pragma unroll
    for (int i = 0; i < BM; i++)
#pragma unroll
      for (int j = 0; j < BN; j++)
        acc[i][j] = __builtin_amdgcn_wmma_f32_16x16x32_bf16(
            false, a[i].v, false, b[j].v, (short)0, acc[i][j], false, false);

    asm volatile("s_wait_asynccnt 0x0" ::: "memory");
    __syncthreads();
  }

  float* base = scores + ((size_t)g << 20);
#pragma unroll
  for (int i = 0; i < BM; i++)
#pragma unroll
    for (int j = 0; j < BN; j++)
      for (int r = 0; r < 8; r++) {
        int o = bm * 128 + wm * 64 + i * 16 + r + 8 * kh;  // output channel
        int s = bn * 128 + wn * 32 + j * 16 + row;         // spatial
        float inv = rsqrtf(bnv[o] + 1e-5f);
        float val = (acc[i][j][r] - bnm[o]) * (inv * bng[o]) + bnb[o];
        val = val / (1.0f + __expf(-val));  // SiLU
        int n = ((o >> 5) << 5) + (s >> 5);
        int m = ((o & 31) << 5) + (s & 31);
        base[(size_t)n * N_TOK + m] = val;
      }
}

// ---------------------------------------------------------------------------
// Softmax over m (1024), one wave per row, bf16 output
// ---------------------------------------------------------------------------
__global__ void k_softmax(const float* __restrict__ sc,
                          unsigned short* __restrict__ attn) {
  int wave = threadIdx.x >> 5, lane = threadIdx.x & 31;
  int row = blockIdx.x * (blockDim.x >> 5) + wave;
  if (row >= GBATCH * N_TOK) return;
  const float* p = sc + (size_t)row * N_TOK;
  float v[32];
  float mx = -3.4e38f;
#pragma unroll
  for (int t = 0; t < 32; t++) { v[t] = p[t * 32 + lane]; mx = fmaxf(mx, v[t]); }
  for (int o = 16; o >= 1; o >>= 1) mx = fmaxf(mx, __shfl_xor(mx, o, 32));
  float s = 0.f;
#pragma unroll
  for (int t = 0; t < 32; t++) { v[t] = __expf(v[t] - mx); s += v[t]; }
  for (int o = 16; o >= 1; o >>= 1) s += __shfl_xor(s, o, 32);
  float inv = 1.0f / s;
  unsigned short* q = attn + (size_t)row * N_TOK;
#pragma unroll
  for (int t = 0; t < 32; t++) q[t * 32 + lane] = f2bf(v[t] * inv);
}

// ---------------------------------------------------------------------------
// out = attn @ v : [1024 x 1024] x [1024 x 64] per g -> o2[token][inner] bf16
// ---------------------------------------------------------------------------
__global__ void k_gemm_av(const unsigned short* __restrict__ attn,
                          const unsigned short* __restrict__ vT,
                          unsigned short* __restrict__ o2) {
  int w = blockIdx.x * (blockDim.x >> 5) + (threadIdx.x >> 5);
  if (w >= GBATCH * 16 * 2) return;
  int g = w >> 5, rem = w & 31, bm = rem >> 1, bn = rem & 1;
  v8f acc[BM][BN] = {};
  wmma_block<BM, BN>(attn + ((size_t)g << 20) + (size_t)bm * 64 * N_TOK, N_TOK,
                     vT + ((size_t)g * DHX + bn * 32) * N_TOK, N_TOK, N_TOK, acc);
  int lane = threadIdx.x & 31, row = lane & 15, kh = lane >> 4;
  int bp = g >> 3, h = g & 7;
#pragma unroll
  for (int i = 0; i < BM; i++)
#pragma unroll
    for (int j = 0; j < BN; j++)
      for (int r = 0; r < 8; r++) {
        int n = bm * 64 + i * 16 + r + 8 * kh;
        int d = bn * 32 + j * 16 + row;
        o2[((size_t)((bp << 10) | n)) * INNERX + (h << 6) + d] = f2bf(acc[i][j][r]);
      }
}

// ---------------------------------------------------------------------------
// Output projection: [4096 x 512] x [512 x 256] + bias -> f32 d_out
// ---------------------------------------------------------------------------
__global__ void k_gemm_out(const unsigned short* __restrict__ o2,
                           const unsigned short* __restrict__ woutT,
                           const float* __restrict__ bout,
                           float* __restrict__ out) {
  int w = blockIdx.x * (blockDim.x >> 5) + (threadIdx.x >> 5);
  if (w >= 64 * 8) return;
  int bm = w >> 3, bn = w & 7;
  v8f acc[BM][BN] = {};
  wmma_block<BM, BN>(o2 + (size_t)bm * 64 * INNERX, INNERX,
                     woutT + (size_t)bn * 32 * INNERX, INNERX, INNERX, acc);
  int lane = threadIdx.x & 31, row = lane & 15, kh = lane >> 4;
#pragma unroll
  for (int i = 0; i < BM; i++)
#pragma unroll
    for (int j = 0; j < BN; j++)
      for (int r = 0; r < 8; r++) {
        int M = bm * 64 + i * 16 + r + 8 * kh;
        int c = bn * 32 + j * 16 + row;
        out[(size_t)M * DIMX + c] = acc[i][j][r] + bout[c];
      }
}

// ---------------------------------------------------------------------------
extern "C" void kernel_launch(void* const* d_in, const int* in_sizes, int n_in,
                              void* d_out, int out_size, void* d_ws, size_t ws_size,
                              hipStream_t stream) {
  const float* x       = (const float*)d_in[0];
  const float* ln_g    = (const float*)d_in[1];
  const float* ln_b    = (const float*)d_in[2];
  const float* w_qkv   = (const float*)d_in[3];
  const float* w_conv1 = (const float*)d_in[4];
  const float* w_conv2 = (const float*)d_in[5];
  const float* w_1x1   = (const float*)d_in[6];
  const float* bn_g    = (const float*)d_in[7];
  const float* bn_b    = (const float*)d_in[8];
  const float* bn_m    = (const float*)d_in[9];
  const float* bn_v    = (const float*)d_in[10];
  const float* w_out   = (const float*)d_in[11];
  const float* b_out   = (const float*)d_in[12];
  float* out = (float*)d_out;

  char* ws = (char*)d_ws;
  unsigned short* wqkvT = (unsigned short*)(ws);                 // 1536x256 bf16
  unsigned short* woutT = (unsigned short*)(ws + 786432);        // 256x512
  unsigned short* w1x1b = (unsigned short*)(ws + 1048576);       // 1024x1024
  unsigned short* xn    = (unsigned short*)(ws + 3145728);       // 4096x256
  unsigned short* qb    = (unsigned short*)(ws + 5242880);       // 32x1024x64
  unsigned short* kb    = (unsigned short*)(ws + 9437184);       // 32x1024x64
  unsigned short* vT    = (unsigned short*)(ws + 13631488);      // 32x64x1024
  unsigned short* o2    = (unsigned short*)(ws + 17825792);      // 4096x512
  float*          dots  = (float*)(ws + 22020096);               // 32x1024x1024 f32
  unsigned short* tb    = (unsigned short*)(ws + 156237824);     // 32x1024x1024 bf16
  unsigned short* attn  = tb;       // reuse conv buffer for attn probs
  float*          scr   = dots;     // reuse dots buffer for post-BN scores

  // 1) weight prep
  k_transpose_bf16<<<(256 * 1536 + 255) / 256, 256, 0, stream>>>(wqkvT, w_qkv, 256, 1536);
  k_transpose_bf16<<<(512 * 256 + 255) / 256, 256, 0, stream>>>(woutT, w_out, 512, 256);
  k_convert_bf16<<<(1024 * 1024 + 255) / 256, 256, 0, stream>>>(w1x1b, w_1x1, 1024 * 1024);

  // 2) layernorm -> xn bf16
  k_layernorm<<<4096 / 8, 256, 0, stream>>>(x, ln_g, ln_b, xn);

  // 3) qkv GEMM (64x48 blocks of 64x32)
  k_gemm_qkv<<<(64 * 48 + 7) / 8, 256, 0, stream>>>(xn, wqkvT, qb, kb, vT);

  // 4) dots = q k^T (32 g * 16x32 blocks)
  k_gemm_dots<<<(GBATCH * 16 * 32 + 7) / 8, 256, 0, stream>>>(qb, kb, dots);

  // 5) fused depthwise convs -> tb (transposed conv image, bf16)
  k_conv<<<(GBATCH * CCH * HWX * HWX) / 256, 256, 0, stream>>>(dots, w_conv1, w_conv2, tb);

  // 6) 1x1 conv GEMM + BN + SiLU (LDS-tiled, async loads): 32g * 8x8 blocks
  k_gemm_mix<<<GBATCH * 8 * 8, 256, 0, stream>>>(w1x1b, tb, bn_g, bn_b, bn_m, bn_v, scr);

  // 7) softmax -> attn bf16
  k_softmax<<<(GBATCH * N_TOK) / 8, 256, 0, stream>>>(scr, attn);

  // 8) attn @ v -> o2
  k_gemm_av<<<(GBATCH * 16 * 2 + 7) / 8, 256, 0, stream>>>(attn, vT, o2);

  // 9) out projection + bias -> f32 out
  k_gemm_out<<<(64 * 8 + 7) / 8, 256, 0, stream>>>(o2, woutT, b_out, out);

  (void)in_sizes; (void)n_in; (void)out_size; (void)ws_size;
}
